// TenVAD_4458176053260
// MI455X (gfx1250) — matrix operand order repaired and checked by reading, not compiled, not run on hardware
//
#include <hip/hip_runtime.h>
#include <hip/hip_bf16.h>

// ---------------------------------------------------------------------------
// TenVAD forward on MI455X (gfx1250, wave32, WMMA).
//  - conv front-end: plain VALU kernels (tiny)
//  - xg GEMMs, LSTM step GEMMs, FC head: v_wmma_f32_16x16x32_f16
//  - LSTM scan: one 16-wave persistent workgroup, H in LDS, Whh + C in VGPRs,
//    next-step xg prefetched (global_prefetch_b8)
// ---------------------------------------------------------------------------

typedef _Float16 half8 __attribute__((ext_vector_type(8)));
typedef _Float16 v16h  __attribute__((ext_vector_type(16)));
typedef float    v8f   __attribute__((ext_vector_type(8)));

#define NB   128      // batch
#define NCH  16       // conv channels
#define NL   2048     // seq length after pool
#define NT   2044     // LSTM time steps (NL - WIN + 1)
#define NTP  2048     // padded time steps
#define NG   256      // gate width (4*64)
#define NH   64       // hidden
#define KSEQ 96       // 80 padded to 3*32

__device__ __forceinline__ float sigm(float x){ return 1.0f/(1.0f + __expf(-x)); }

__device__ __forceinline__ v16h frag_join(half8 lo, half8 hi){
  v16h a;
#pragma unroll
  for (int i=0;i<8;i++){ a[i]=lo[i]; a[i+8]=hi[i]; }
  return a;
}

// A operand (16x32 f16): lane l<16 holds row l, K {0..7,16..23}; lane l+16: K {8..15,24..31}
// 'rowk' points at element [row][kchunk*32] of a row-major [M][K] tile.
__device__ __forceinline__ v16h afrag(const _Float16* rowk, int lane){
  int off = (lane & 16) ? 8 : 0;
  half8 lo = *(const half8*)(rowk + off);
  half8 hi = *(const half8*)(rowk + off + 16);
  return frag_join(lo, hi);
}
// B operand (32x16 f16): lane l<16 holds col l, K {0..7,16..23}; lane l+16: K {8..15,24..31}.
// 'colk' points at element [n][kchunk*32] of an [N][K] (row-major over N) tile.
__device__ __forceinline__ v16h bfrag(const _Float16* colk, int lane){
  int off = (lane & 16) ? 8 : 0;
  half8 lo = *(const half8*)(colk + off);
  half8 hi = *(const half8*)(colk + off + 16);
  return frag_join(lo, hi);
}

__device__ __forceinline__ v8f wmma16(v16h a, v16h b, v8f c){
  return __builtin_amdgcn_wmma_f32_16x16x32_f16(false, a, false, b, (short)0, c, false, false);
}

// load 8 consecutive f16 and widen to v8f (xg accumulator init)
__device__ __forceinline__ v8f ldx8f(const _Float16* p){
  half8 xv = *(const half8*)p;
  v8f c;
#pragma unroll
  for (int r=0;r<8;r++) c[r] = (float)xv[r];
  return c;
}

// ------------------------- weight prep (f32 -> f16) -------------------------
__global__ void vad_prep(const float* Wih1, const float* Whh1,
                         const float* Wih2, const float* Whh2,
                         const float* fc1w,
                         const float* bih1, const float* bhh1,
                         const float* bih2, const float* bhh2,
                         _Float16* Wih1h, _Float16* Whh1h,
                         _Float16* Wih2h, _Float16* Whh2h,
                         _Float16* fc1wh, float* bsum1, float* bsum2)
{
  int tid = threadIdx.x + blockIdx.x * blockDim.x;
  int nth = blockDim.x * gridDim.x;
  for (int i = tid; i < NG*KSEQ; i += nth){
    int g = i / KSEQ, k = i % KSEQ;
    Wih1h[i] = (k < 80) ? (_Float16)Wih1[g*80 + k] : (_Float16)0.0f;
  }
  for (int i = tid; i < NG*NH; i += nth) Whh1h[i] = (_Float16)Whh1[i];
  for (int i = tid; i < NG*NH; i += nth) Wih2h[i] = (_Float16)Wih2[i];
  for (int i = tid; i < NG*NH; i += nth) Whh2h[i] = (_Float16)Whh2[i];
  for (int i = tid; i < 32*128; i += nth) fc1wh[i] = (_Float16)fc1w[i];
  for (int i = tid; i < NG; i += nth){
    bsum1[i] = bih1[i] + bhh1[i];
    bsum2[i] = bih2[i] + bhh2[i];
  }
}

// ------------------- front conv + pointwise + relu + pool -------------------
__global__ void vad_front(const float* __restrict__ x, const float* __restrict__ dw,
                          const float* __restrict__ pw, const float* __restrict__ pb,
                          float* __restrict__ p)
{
  int b  = blockIdx.y;
  int l2 = blockIdx.x * blockDim.x + threadIdx.x;   // 0..2047
  float w[3][3];
#pragma unroll
  for (int i=0;i<3;i++)
#pragma unroll
    for (int j=0;j<3;j++) w[i][j] = dw[i*3 + j];
  const float* xb = x + (size_t)b * 3 * 4096;
  float y[2];
#pragma unroll
  for (int s=0;s<2;s++){
    int l = 2*l2 + s;
    float acc = 0.0f;
#pragma unroll
    for (int i=0;i<3;i++){
      const float* xr = xb + i*4096;
#pragma unroll
      for (int j=0;j<3;j++){
        int li = l + j - 1;
        float v = (li >= 0 && li < 4096) ? xr[li] : 0.0f;
        acc += v * w[i][j];
      }
    }
    y[s] = acc;
  }
#pragma unroll
  for (int ch=0; ch<NCH; ch++){
    float a = fmaxf(y[0]*pw[ch] + pb[ch], 0.0f);
    float c = fmaxf(y[1]*pw[ch] + pb[ch], 0.0f);
    p[((size_t)b*NCH + ch)*NL + l2] = fmaxf(a, c);
  }
}

// ---------------- separable conv (depthwise3 + pointwise16x16) --------------
__global__ void vad_sep(const float* __restrict__ in, const float* __restrict__ dw,
                        const float* __restrict__ pw, const float* __restrict__ pb,
                        float* __restrict__ out, _Float16* __restrict__ outh)
{
  __shared__ float s[NCH][130];
  int b   = blockIdx.y;
  int l0  = blockIdx.x * 128;
  int tid = threadIdx.x;                 // 128 threads
  for (int i = tid; i < NCH*130; i += 128){
    int ch = i / 130, pos = i % 130;
    int gl = l0 + pos - 1;
    s[ch][pos] = (gl >= 0 && gl < NL) ? in[((size_t)b*NCH + ch)*NL + gl] : 0.0f;
  }
  __syncthreads();
  float dv[NCH];
#pragma unroll
  for (int c=0;c<NCH;c++)
    dv[c] = s[c][tid]*dw[c*3] + s[c][tid+1]*dw[c*3+1] + s[c][tid+2]*dw[c*3+2];
  int l = l0 + tid;
#pragma unroll
  for (int o=0;o<NCH;o++){
    float acc = pb[o];
#pragma unroll
    for (int c=0;c<NCH;c++) acc += dv[c]*pw[o*NCH + c];
    acc = fmaxf(acc, 0.0f);
    out[((size_t)b*NCH + o)*NL + l] = acc;
    if (outh) outh[((size_t)b*NCH + o)*NL + l] = (_Float16)acc;
  }
}

// ------------ GEMM: xg1[t][g][b] = seq(b,t,:) @ Wih1^T + bih+bhh -----------
// block: one batch b, 64 consecutive t.  8 waves; wave w -> M-tile w/2, N-half w%2.
__global__ void __launch_bounds__(256)
vad_gemm_seq(const _Float16* __restrict__ fh,   // [B][16][2048]
             const _Float16* __restrict__ Wh,   // [256][96]
             const float* __restrict__ bias,    // [256]
             _Float16* __restrict__ xg)         // [NTP][256][128]
{
  __shared__ _Float16 sW[NG][KSEQ];   // 48 KB
  __shared__ _Float16 sA[64][KSEQ];   // 12 KB
  int tid = threadIdx.x, lane = tid & 31, wid = tid >> 5;
  int b  = blockIdx.y;
  int t0 = blockIdx.x * 64;
  for (int i = tid; i < NG*KSEQ; i += 256) ((_Float16*)sW)[i] = Wh[i];
  for (int i = tid; i < 64*KSEQ; i += 256){
    int m = i / KSEQ, k = i % KSEQ;
    _Float16 v = (_Float16)0.0f;
    if (k < 80){
      int c = k / 5, w = k % 5;
      int tt = t0 + m + w; if (tt > NL-1) tt = NL-1;
      v = fh[((size_t)b*NCH + c)*NL + tt];
    }
    sA[m][k] = v;
  }
  __syncthreads();
  int mt  = wid >> 1;
  int nb  = (wid & 1) * 8;
  int col = lane & 15;
  int rb  = (lane & 16) ? 8 : 0;
  for (int nt = 0; nt < 8; ++nt){
    int n0 = (nb + nt) * 16;
    v8f acc = {};
#pragma unroll
    for (int kc = 0; kc < 3; ++kc){
      v16h a = afrag(&sA[mt*16 + col][kc*32], lane);
      v16h w = bfrag(&sW[n0 + col][kc*32], lane);
      acc = wmma16(a, w, acc);
    }
    int g = n0 + col;
    float bb = bias[g];
#pragma unroll
    for (int r = 0; r < 8; ++r){
      int t = t0 + mt*16 + rb + r;
      xg[((size_t)t*NG + g)*NB + b] = (_Float16)(acc[r] + bb);
    }
  }
}

// -------- GEMM: xg2[t][g][b] = x1[t][b][:] @ Wih2^T + bih2+bhh2 ------------
__global__ void __launch_bounds__(256)
vad_gemm_x(const _Float16* __restrict__ x1,    // [NTP][128][64]
           const _Float16* __restrict__ Wh,    // [256][64]
           const float* __restrict__ bias,     // [256]
           _Float16* __restrict__ xg)          // [NTP][256][128]
{
  __shared__ _Float16 sW[NG][NH];    // 32 KB
  int tid = threadIdx.x, lane = tid & 31, wid = tid >> 5;
  int b  = blockIdx.y;
  int t0 = blockIdx.x * 64;
  for (int i = tid; i < NG*NH; i += 256) ((_Float16*)sW)[i] = Wh[i];
  __syncthreads();
  int mt  = wid >> 1;
  int nb  = (wid & 1) * 8;
  int col = lane & 15;
  int rb  = (lane & 16) ? 8 : 0;
  int trow = t0 + mt*16 + col;                     // A row for this lane
  const _Float16* arow = x1 + ((size_t)trow*NB + b)*NH;
  v16h a0 = afrag(arow,      lane);
  v16h a1 = afrag(arow + 32, lane);
  for (int nt = 0; nt < 8; ++nt){
    int n0 = (nb + nt) * 16;
    v8f acc = {};
    acc = wmma16(a0, bfrag(&sW[n0 + col][0],  lane), acc);
    acc = wmma16(a1, bfrag(&sW[n0 + col][32], lane), acc);
    int g = n0 + col;
    float bb = bias[g];
#pragma unroll
    for (int r = 0; r < 8; ++r){
      int t = t0 + mt*16 + rb + r;
      xg[((size_t)t*NG + g)*NB + b] = (_Float16)(acc[r] + bb);
    }
  }
}

// ----------------------------- LSTM scan -----------------------------------
// Single workgroup, 512 threads = 16 waves (4/SIMD).  Wave (mt2,jj) owns batch
// rows mt2*32..mt2*32+31 (two 16-row M-tiles) and hidden cols jj*16..jj*16+15.
// Whh fragments (8 x v16h = 64 VGPRs) are hoisted out of the t-loop; H state
// lives in LDS (f16); C state lives in accumulator-side VGPRs.  Per step:
// 8 LDS b128 loads (A), 8 global b128 loads (xg), 16 WMMAs, 2 barriers, plus
// global_prefetch of the next step's xg fragments.
__global__ void __launch_bounds__(512)
vad_lstm_scan(const _Float16* __restrict__ xg,   // [NTP][256][128]
              const _Float16* __restrict__ Whh,  // [256][64]
              const float* __restrict__ h0,      // [128][64]
              const float* __restrict__ c0,      // [128][64]
              _Float16* __restrict__ xout,       // [NTP][128][64]
              float* __restrict__ hN, float* __restrict__ cN)
{
  __shared__ _Float16 sH[NB][NH];   // 16 KB
  int tid = threadIdx.x, lane = tid & 31, wid = tid >> 5;
  for (int i = tid; i < NB*NH; i += 512) ((_Float16*)sH)[i] = (_Float16)h0[i];
  int mt2 = wid >> 2;                 // 0..3  -> M-tiles 2*mt2, 2*mt2+1
  int jj  = wid & 3;                  // 0..3
  int col = lane & 15;
  int rb  = (lane & 16) ? 8 : 0;
  int n   = jj*16 + col;              // hidden column
  int ma  = (mt2*2)    * 16;          // first M-tile row base
  int mb  = (mt2*2 + 1)* 16;
  int m0a = ma + rb;                  // this lane's first fragment row (tile A)
  int m0b = mb + rb;                  // this lane's first fragment row (tile B)

  // hoist Whh fragments: gate gi, k-chunk kc
  v16h bw[4][2];
#pragma unroll
  for (int gi = 0; gi < 4; ++gi){
    int gn = gi*64 + n;
#pragma unroll
    for (int kc = 0; kc < 2; ++kc)
      bw[gi][kc] = bfrag(Whh + gn*NH + kc*32, lane);
  }

  float csta[8], cstb[8];
#pragma unroll
  for (int r = 0; r < 8; ++r){
    csta[r] = c0[(m0a + r)*NH + n];
    cstb[r] = c0[(m0b + r)*NH + n];
  }
  __syncthreads();

  for (int t = 0; t < NT; ++t){
    const _Float16* xgt = xg + (size_t)t*NG*NB;
    // A fragments for both M-tiles (all reads of sH happen here / in WMMA)
    v16h a00 = afrag(&sH[ma + col][0],  lane);
    v16h a01 = afrag(&sH[ma + col][32], lane);
    v16h a10 = afrag(&sH[mb + col][0],  lane);
    v16h a11 = afrag(&sH[mb + col][32], lane);
    v8f acca[4], accb[4];
#pragma unroll
    for (int gi = 0; gi < 4; ++gi){
      int gn = gi*64 + n;
      v8f ca = ldx8f(xgt + gn*NB + m0a);
      v8f cb = ldx8f(xgt + gn*NB + m0b);
      ca = wmma16(a00, bw[gi][0], ca);
      ca = wmma16(a01, bw[gi][1], ca);
      cb = wmma16(a10, bw[gi][0], cb);
      cb = wmma16(a11, bw[gi][1], cb);
      acca[gi] = ca;
      accb[gi] = cb;
    }
    // prefetch next step's xg fragments into the WGP cache (no counter cost)
    if (t + 1 < NT){
      const _Float16* xgn = xgt + (size_t)NG*NB;
#pragma unroll
      for (int gi = 0; gi < 4; ++gi){
        int gn = gi*64 + n;
        __builtin_prefetch(xgn + gn*NB + m0a, 0, 1);
        __builtin_prefetch(xgn + gn*NB + m0b, 0, 1);
      }
    }
    __syncthreads();                          // all reads of sH complete
    _Float16* xo = xout + (size_t)t*NB*NH;
#pragma unroll
    for (int r = 0; r < 8; ++r){
      // tile A
      {
        int m = m0a + r;
        float iv = sigm(acca[0][r]);
        float fv = sigm(acca[1][r]);
        float gv = tanhf(acca[2][r]);
        float ov = sigm(acca[3][r]);
        float cc = fv*csta[r] + iv*gv;
        csta[r] = cc;
        float hh = ov*tanhf(cc);
        sH[m][n] = (_Float16)hh;
        xo[m*NH + n] = (_Float16)hh;
        if (t == NT-1){ hN[m*NH + n] = hh; cN[m*NH + n] = cc; }
      }
      // tile B
      {
        int m = m0b + r;
        float iv = sigm(accb[0][r]);
        float fv = sigm(accb[1][r]);
        float gv = tanhf(accb[2][r]);
        float ov = sigm(accb[3][r]);
        float cc = fv*cstb[r] + iv*gv;
        cstb[r] = cc;
        float hh = ov*tanhf(cc);
        sH[m][n] = (_Float16)hh;
        xo[m*NH + n] = (_Float16)hh;
        if (t == NT-1){ hN[m*NH + n] = hh; cN[m*NH + n] = cc; }
      }
    }
    __syncthreads();                          // new H visible
  }
}

// --------------------------- FC head ---------------------------------------
// z = relu([x1|x2] @ fc1^T + b1); out = sigmoid(z @ fc2^T + b2)
// block: one batch b, 128 t; 8 waves, one 16-t M-tile each. N=32 (two tiles).
__global__ void __launch_bounds__(256)
vad_fc(const _Float16* __restrict__ x1, const _Float16* __restrict__ x2,
       const _Float16* __restrict__ w1h,   // [32][128]
       const float* __restrict__ b1f, const float* __restrict__ w2,
       const float* __restrict__ b2f, float* __restrict__ out)
{
  __shared__ _Float16 sW[32][128];   // 8 KB
  int tid = threadIdx.x, lane = tid & 31, wid = tid >> 5;
  for (int i = tid; i < 32*128; i += 256) ((_Float16*)sW)[i] = w1h[i];
  __syncthreads();
  int b   = blockIdx.y;
  int t0  = blockIdx.x * 128 + wid * 16;
  int col = lane & 15;
  int rb  = (lane & 16) ? 8 : 0;
  int trow = t0 + col;
  const _Float16* r1 = x1 + ((size_t)trow*NB + b)*NH;
  const _Float16* r2 = x2 + ((size_t)trow*NB + b)*NH;
  v8f za = {}, zb = {};
#pragma unroll
  for (int kc = 0; kc < 4; ++kc){
    const _Float16* src = (kc < 2) ? (r1 + kc*32) : (r2 + (kc-2)*32);
    v16h a  = afrag(src, lane);
    za = wmma16(a, bfrag(&sW[col][kc*32],      lane), za);
    zb = wmma16(a, bfrag(&sW[16 + col][kc*32], lane), zb);
  }
  float biasA = b1f[col],  biasB = b1f[16 + col];
  float w2A   = w2[col],   w2B   = w2[16 + col];
  float bout  = b2f[0];
#pragma unroll
  for (int r = 0; r < 8; ++r){
    float zA = fmaxf(za[r] + biasA, 0.0f);
    float zB = fmaxf(zb[r] + biasB, 0.0f);
    float s  = zA*w2A + zB*w2B;
    s += __shfl_xor(s, 1, 16);
    s += __shfl_xor(s, 2, 16);
    s += __shfl_xor(s, 4, 16);
    s += __shfl_xor(s, 8, 16);
    if (col == 0){
      int t = t0 + rb + r;
      if (t < NT) out[(size_t)b*NT + t] = sigm(s + bout);
    }
  }
}

// ---------------------------------------------------------------------------
extern "C" void kernel_launch(void* const* d_in, const int* in_sizes, int n_in,
                              void* d_out, int out_size, void* d_ws, size_t ws_size,
                              hipStream_t stream)
{
  const float* x    = (const float*)d_in[0];
  const float* h1in = (const float*)d_in[1];
  const float* c1in = (const float*)d_in[2];
  const float* h2in = (const float*)d_in[3];
  const float* c2in = (const float*)d_in[4];
  const float* cdw  = (const float*)d_in[5];
  const float* cpw  = (const float*)d_in[6];
  const float* cpb  = (const float*)d_in[7];
  const float* s1dw = (const float*)d_in[8];
  const float* s1pw = (const float*)d_in[9];
  const float* s1pb = (const float*)d_in[10];
  const float* s2dw = (const float*)d_in[11];
  const float* s2pw = (const float*)d_in[12];
  const float* s2pb = (const float*)d_in[13];
  const float* Wih1 = (const float*)d_in[14];
  const float* Whh1 = (const float*)d_in[15];
  const float* bih1 = (const float*)d_in[16];
  const float* bhh1 = (const float*)d_in[17];
  const float* Wih2 = (const float*)d_in[18];
  const float* Whh2 = (const float*)d_in[19];
  const float* bih2 = (const float*)d_in[20];
  const float* bhh2 = (const float*)d_in[21];
  const float* fc1w = (const float*)d_in[22];
  const float* fc1b = (const float*)d_in[23];
  const float* fc2w = (const float*)d_in[24];
  const float* fc2b = (const float*)d_in[25];

  // workspace carve-up
  char* w = (char*)d_ws;
  auto alloc = [&](size_t bytes)->char*{
    char* r = w; w += (bytes + 255) & ~(size_t)255; return r;
  };
  float*    pbuf  = (float*)   alloc((size_t)NB*NCH*NL*4);
  float*    qbuf  = (float*)   alloc((size_t)NB*NCH*NL*4);
  _Float16* fh    = (_Float16*)alloc((size_t)NB*NCH*NL*2);
  _Float16* Wih1h = (_Float16*)alloc((size_t)NG*KSEQ*2);
  _Float16* Whh1h = (_Float16*)alloc((size_t)NG*NH*2);
  _Float16* Wih2h = (_Float16*)alloc((size_t)NG*NH*2);
  _Float16* Whh2h = (_Float16*)alloc((size_t)NG*NH*2);
  _Float16* fc1wh = (_Float16*)alloc((size_t)32*128*2);
  float*    bsum1 = (float*)   alloc((size_t)NG*4);
  float*    bsum2 = (float*)   alloc((size_t)NG*4);
  _Float16* xg1   = (_Float16*)alloc((size_t)NTP*NG*NB*2);
  _Float16* xg2   = (_Float16*)alloc((size_t)NTP*NG*NB*2);
  _Float16* x1b   = (_Float16*)alloc((size_t)NTP*NB*NH*2);
  _Float16* x2b   = (_Float16*)alloc((size_t)NTP*NB*NH*2);

  float* out = (float*)d_out;                 // [128][2044]
  float* h1n = out + (size_t)NB*NT;
  float* c1n = h1n + NB*NH;
  float* h2n = c1n + NB*NH;
  float* c2n = h2n + NB*NH;

  vad_prep<<<8, 256, 0, stream>>>(Wih1, Whh1, Wih2, Whh2, fc1w,
                                  bih1, bhh1, bih2, bhh2,
                                  Wih1h, Whh1h, Wih2h, Whh2h,
                                  fc1wh, bsum1, bsum2);

  vad_front<<<dim3(NL/256, NB), 256, 0, stream>>>(x, cdw, cpw, cpb, pbuf);
  vad_sep  <<<dim3(NL/128, NB), 128, 0, stream>>>(pbuf, s1dw, s1pw, s1pb, qbuf, ((_Float16*)nullptr));
  vad_sep  <<<dim3(NL/128, NB), 128, 0, stream>>>(qbuf, s2dw, s2pw, s2pb, pbuf, fh);

  vad_gemm_seq<<<dim3(NL/64, NB), 256, 0, stream>>>(fh, Wih1h, bsum1, xg1);
  vad_lstm_scan<<<1, 512, 0, stream>>>(xg1, Whh1h, h1in, c1in, x1b, h1n, c1n);
  vad_gemm_x  <<<dim3(NL/64, NB), 256, 0, stream>>>(x1b, Wih2h, bsum2, xg2);
  vad_lstm_scan<<<1, 512, 0, stream>>>(xg2, Whh2h, h2in, c2in, x2b, h2n, c2n);

  vad_fc<<<dim3(NL/128, NB), 256, 0, stream>>>(x1b, x2b, fc1wh, fc1b, fc2w, fc2b, out);
}
